// AttnDecoderRNN_52991306498250
// MI455X (gfx1250) — compile-verified, matrix operand order checked by reading, not run on hardware
//
#include <hip/hip_runtime.h>
#include <hip/hip_bf16.h>
#include <math.h>

#define HDIM 512
#define VDIM 50257
#define BDIM 128
#define SDIM 10
#define TDIM 10

typedef __bf16 bf16_t;
typedef __attribute__((ext_vector_type(16))) __bf16 v16bf;
typedef __attribute__((ext_vector_type(8)))  __bf16 v8bf;
typedef __attribute__((ext_vector_type(8)))  float  v8f;

// ---------------------------------------------------------------------------
// fp32 -> bf16 conversion (weights converted once per launch; L2-resident after)
// ---------------------------------------------------------------------------
__global__ void k_f32_to_bf16(const float* __restrict__ src, bf16_t* __restrict__ dst, size_t n) {
    size_t i = (size_t)blockIdx.x * blockDim.x + threadIdx.x;
    if (i < n) dst[i] = (bf16_t)src[i];
}

__global__ void k_init_h(const float* __restrict__ eh, float* __restrict__ h,
                         bf16_t* __restrict__ hb, int n) {
    int i = blockIdx.x * blockDim.x + threadIdx.x;
    if (i < n) { float v = eh[i]; h[i] = v; hb[i] = (bf16_t)v; }
}

__global__ void k_copy_f32(const float* __restrict__ src, float* __restrict__ dst, int n) {
    int i = blockIdx.x * blockDim.x + threadIdx.x;
    if (i < n) dst[i] = src[i];
}

// Helper: pack two contiguous v8bf loads into the v16bf WMMA operand.
__device__ __forceinline__ v16bf load_frag(const bf16_t* p) {
    v8bf lo = *(const v8bf*)(p);
    v8bf hi = *(const v8bf*)(p + 16);
    v16bf v;
#pragma unroll
    for (int i = 0; i < 8; ++i) { v[i] = lo[i]; v[i + 8] = hi[i]; }
    return v;
}

// ---------------------------------------------------------------------------
// Small-GEMM kernel: one wave per 16x16 tile, 8 waves/block over 8 n-tiles.
// C[M,N] = A[M,K] @ W[N,K]^T + bias[N].  grid = (ceil(Ntiles/8), M/16).
// Used for q / gi / gh / Ua_keys where weights are tiny and parallelism wins.
// ---------------------------------------------------------------------------
__global__ void k_wmma_gemm(const bf16_t* __restrict__ A, int lda,
                            const bf16_t* __restrict__ W, int ldw,
                            const float* __restrict__ bias,
                            float* __restrict__ C, long long ldc,
                            int N, int K)
{
    const int lane = threadIdx.x & 31;
    const int wave = threadIdx.x >> 5;
    const int hi   = lane >> 4;
    const int r    = lane & 15;

    const int mt = blockIdx.y;
    const int nt = blockIdx.x * 8 + wave;
    if (nt * 16 >= N) return;                   // uniform per wave

    const int m = mt * 16 + r;
    const int n = nt * 16 + r;
    const bf16_t* pA = A + (size_t)m * lda;
    const bf16_t* pW = W + (size_t)(n < N ? n : 0) * ldw;

    v8f acc = {};
    for (int kt = 0; kt < K; kt += 32) {
        const int k = kt + hi * 8;
        v16bf av = load_frag(pA + k);
        v16bf bv = load_frag(pW + k);
        acc = __builtin_amdgcn_wmma_f32_16x16x32_bf16(false, av, false, bv,
                                                      (short)0, acc, false, false);
    }

    if (n < N) {
        float bvv = bias[n];
#pragma unroll
        for (int rr = 0; rr < 8; ++rr) {
            int mrow = mt * 16 + rr + 8 * hi;
            C[(size_t)mrow * (size_t)ldc + n] = acc[rr] + bvv;
        }
    }
}

// ---------------------------------------------------------------------------
// Weight-streaming GEMM for the huge logits projection (N = 50257).
// One wave owns one n-tile and ALL 8 m-tiles of a 128-row group: each W
// fragment loaded from L2 feeds 8 WMMAs (128 output rows), so out_w is read
// exactly once per step.  grid = (ceil(Ntiles/8), M/128), block = 256.
// ---------------------------------------------------------------------------
__global__ void k_wmma_gemm_mblock(const bf16_t* __restrict__ A, int lda,
                                   const bf16_t* __restrict__ W, int ldw,
                                   const float* __restrict__ bias,
                                   float* __restrict__ C, long long ldc,
                                   int N, int K)
{
    const int lane = threadIdx.x & 31;
    const int wave = threadIdx.x >> 5;
    const int hi   = lane >> 4;
    const int r    = lane & 15;

    const int nt    = blockIdx.x * 8 + wave;
    const int mbase = blockIdx.y * 128;
    if (nt * 16 >= N) return;                   // uniform per wave

    const int n = nt * 16 + r;
    const bf16_t* pW = W + (size_t)(n < N ? n : 0) * ldw;
    const bf16_t* pA = A + (size_t)(mbase + r) * lda;

    v8f acc[8];
#pragma unroll
    for (int mt = 0; mt < 8; ++mt) acc[mt] = {};

    for (int kt = 0; kt < K; kt += 32) {
        const int k = kt + hi * 8;
        v16bf bv = load_frag(pW + k);           // loaded once, used 8x
#pragma unroll
        for (int mt = 0; mt < 8; ++mt) {
            v16bf av = load_frag(pA + (size_t)(mt * 16) * lda + k);
            acc[mt] = __builtin_amdgcn_wmma_f32_16x16x32_bf16(false, av, false, bv,
                                                              (short)0, acc[mt], false, false);
        }
    }

    if (n < N) {
        float bvv = bias[n];
#pragma unroll
        for (int mt = 0; mt < 8; ++mt) {
#pragma unroll
            for (int rr = 0; rr < 8; ++rr) {
                int mrow = mbase + mt * 16 + rr + 8 * hi;
                C[(size_t)mrow * (size_t)ldc + n] = acc[mt][rr] + bvv;
            }
        }
    }
}

// ---------------------------------------------------------------------------
// Attention: scores = va . tanh(q + Ua_keys), softmax over S, ctx = w @ keys.
// Also builds x = [embedding[token] ; ctx] in bf16 for the GRU input GEMM.
// ---------------------------------------------------------------------------
__global__ void k_attention(const float* __restrict__ q,        // [B,H]
                            const float* __restrict__ ua_keys,  // [B,S,H]
                            const float* __restrict__ keys,     // [B,S,H] fp32
                            const float* __restrict__ va_w,     // [H]
                            const float* __restrict__ va_b,     // [1]
                            const float* __restrict__ emb,      // [V,H]
                            const int*   __restrict__ target,   // [B,S]
                            int t,
                            bf16_t* __restrict__ x,             // [B,2H] bf16
                            float*  __restrict__ attn_out)      // [B,T,S]
{
    const int b = blockIdx.x;
    const int tid = threadIdx.x;
    __shared__ float red[256];
    __shared__ float wsh[SDIM];

    float partial[SDIM];
#pragma unroll
    for (int s = 0; s < SDIM; ++s) partial[s] = 0.f;

    for (int h = tid; h < HDIM; h += 256) {
        float qv = q[(size_t)b * HDIM + h];
        float vv = va_w[h];
#pragma unroll
        for (int s = 0; s < SDIM; ++s)
            partial[s] += vv * tanhf(qv + ua_keys[((size_t)b * SDIM + s) * HDIM + h]);
    }
    for (int s = 0; s < SDIM; ++s) {
        red[tid] = partial[s];
        __syncthreads();
        for (int off = 128; off > 0; off >>= 1) {
            if (tid < off) red[tid] += red[tid + off];
            __syncthreads();
        }
        if (tid == 0) wsh[s] = red[0] + va_b[0];
        __syncthreads();
    }
    if (tid == 0) {
        float mx = wsh[0];
        for (int s = 1; s < SDIM; ++s) mx = fmaxf(mx, wsh[s]);
        float sum = 0.f;
        for (int s = 0; s < SDIM; ++s) { wsh[s] = expf(wsh[s] - mx); sum += wsh[s]; }
        float inv = 1.f / sum;
        for (int s = 0; s < SDIM; ++s) {
            wsh[s] *= inv;
            attn_out[((size_t)b * TDIM + t) * SDIM + s] = wsh[s];
        }
    }
    __syncthreads();

    int tok = (t == 0) ? 0 : target[(size_t)b * SDIM + (t - 1)];   // SOS_TOKEN = 0
    for (int h = tid; h < HDIM; h += 256) {
        float ctx = 0.f;
#pragma unroll
        for (int s = 0; s < SDIM; ++s)
            ctx += wsh[s] * keys[((size_t)b * SDIM + s) * HDIM + h];
        x[(size_t)b * (2 * HDIM) + h]        = (bf16_t)emb[(size_t)tok * HDIM + h];
        x[(size_t)b * (2 * HDIM) + HDIM + h] = (bf16_t)ctx;
    }
}

// ---------------------------------------------------------------------------
// GRU gate fusion: h_new = (1-z)*n + z*h   (writes fp32 h and bf16 h in place)
// ---------------------------------------------------------------------------
__global__ void k_gru(const float* __restrict__ gi, const float* __restrict__ gh,
                      float* __restrict__ h, bf16_t* __restrict__ hb)
{
    int idx = blockIdx.x * blockDim.x + threadIdx.x;   // < B*H
    int b = idx / HDIM, j = idx % HDIM;
    const float* gib = gi + (size_t)b * 3 * HDIM;
    const float* ghb = gh + (size_t)b * 3 * HDIM;
    float ir = gib[j], iz = gib[HDIM + j], in_ = gib[2 * HDIM + j];
    float hr = ghb[j], hz = ghb[HDIM + j], hn  = ghb[2 * HDIM + j];
    float r = 1.f / (1.f + expf(-(ir + hr)));
    float z = 1.f / (1.f + expf(-(iz + hz)));
    float n = tanhf(in_ + r * hn);
    float hnew = (1.f - z) * n + z * h[idx];
    h[idx]  = hnew;
    hb[idx] = (bf16_t)hnew;
}

// ---------------------------------------------------------------------------
// In-place log-softmax over V for step t: one block per batch row.
// ---------------------------------------------------------------------------
__global__ void k_log_softmax(float* __restrict__ logp, int t)
{
    const int b = blockIdx.x, tid = threadIdx.x;
    float* row = logp + ((size_t)b * TDIM + t) * VDIM;
    __shared__ float red[256];

    float m = -INFINITY;
    for (int i = tid; i < VDIM; i += 256) m = fmaxf(m, row[i]);
    red[tid] = m; __syncthreads();
    for (int off = 128; off > 0; off >>= 1) {
        if (tid < off) red[tid] = fmaxf(red[tid], red[tid + off]);
        __syncthreads();
    }
    float mx = red[0];
    __syncthreads();

    float s = 0.f;
    for (int i = tid; i < VDIM; i += 256) s += expf(row[i] - mx);
    red[tid] = s; __syncthreads();
    for (int off = 128; off > 0; off >>= 1) {
        if (tid < off) red[tid] += red[tid + off];
        __syncthreads();
    }
    float lse = logf(red[0]) + mx;
    __syncthreads();

    for (int i = tid; i < VDIM; i += 256) row[i] = row[i] - lse;
}

// ---------------------------------------------------------------------------
extern "C" void kernel_launch(void* const* d_in, const int* in_sizes, int n_in,
                              void* d_out, int out_size, void* d_ws, size_t ws_size,
                              hipStream_t stream)
{
    (void)in_sizes; (void)n_in; (void)out_size; (void)ws_size;

    const float* enc_out = (const float*)d_in[0];   // [B,S,H]
    const float* enc_h   = (const float*)d_in[1];   // [1,B,H]
    const int*   target  = (const int*)d_in[2];     // [B,S]
    const float* emb     = (const float*)d_in[3];   // [V,H]
    const float* WaW     = (const float*)d_in[4];   // [H,H]
    const float* WaB     = (const float*)d_in[5];
    const float* UaW     = (const float*)d_in[6];   // [H,H]
    const float* UaB     = (const float*)d_in[7];
    const float* VaW     = (const float*)d_in[8];   // [1,H]
    const float* VaB     = (const float*)d_in[9];
    const float* Wih     = (const float*)d_in[10];  // [3H,2H]
    const float* Whh     = (const float*)d_in[11];  // [3H,H]
    const float* Bih     = (const float*)d_in[12];
    const float* Bhh     = (const float*)d_in[13];
    const float* OutW    = (const float*)d_in[14];  // [V,H]
    const float* OutB    = (const float*)d_in[15];

    float* out_logp = (float*)d_out;                               // [B,T,V]
    float* out_hT   = out_logp + (size_t)BDIM * TDIM * VDIM;       // [1,B,H]
    float* out_attn = out_hT + (size_t)BDIM * HDIM;                // [B,T,S]

    // workspace bump allocator (total ~64 MB)
    char* ws = (char*)d_ws;
    size_t off = 0;
    auto alloc = [&](size_t bytes) -> void* {
        off = (off + 255) & ~(size_t)255;
        void* p = ws + off;
        off += bytes;
        return p;
    };
    bf16_t* outw_b  = (bf16_t*)alloc((size_t)VDIM * HDIM * 2);        // 51.5 MB (L2-resident)
    bf16_t* waw_b   = (bf16_t*)alloc((size_t)HDIM * HDIM * 2);
    bf16_t* uaw_b   = (bf16_t*)alloc((size_t)HDIM * HDIM * 2);
    bf16_t* wih_b   = (bf16_t*)alloc((size_t)3 * HDIM * 2 * HDIM * 2);
    bf16_t* whh_b   = (bf16_t*)alloc((size_t)3 * HDIM * HDIM * 2);
    bf16_t* keys_b  = (bf16_t*)alloc((size_t)BDIM * SDIM * HDIM * 2);
    float*  ua_keys = (float*)alloc((size_t)BDIM * SDIM * HDIM * 4);
    float*  hbuf    = (float*)alloc((size_t)BDIM * HDIM * 4);
    bf16_t* hb      = (bf16_t*)alloc((size_t)BDIM * HDIM * 2);
    float*  qbuf    = (float*)alloc((size_t)BDIM * HDIM * 4);
    bf16_t* xbuf    = (bf16_t*)alloc((size_t)BDIM * 2 * HDIM * 2);
    float*  gibuf   = (float*)alloc((size_t)BDIM * 3 * HDIM * 4);
    float*  ghbuf   = (float*)alloc((size_t)BDIM * 3 * HDIM * 4);

    auto cvt = [&](const float* s, bf16_t* dst, size_t n) {
        unsigned g = (unsigned)((n + 255) / 256);
        k_f32_to_bf16<<<g, 256, 0, stream>>>(s, dst, n);
    };
    cvt(OutW, outw_b, (size_t)VDIM * HDIM);
    cvt(WaW,  waw_b,  (size_t)HDIM * HDIM);
    cvt(UaW,  uaw_b,  (size_t)HDIM * HDIM);
    cvt(Wih,  wih_b,  (size_t)3 * HDIM * 2 * HDIM);
    cvt(Whh,  whh_b,  (size_t)3 * HDIM * HDIM);
    cvt(enc_out, keys_b, (size_t)BDIM * SDIM * HDIM);
    k_init_h<<<(BDIM * HDIM) / 256, 256, 0, stream>>>(enc_h, hbuf, hb, BDIM * HDIM);

    // Ua_keys = keys @ UaW^T + UaB : M = B*S = 1280, N = 512, K = 512
    k_wmma_gemm<<<dim3((HDIM / 16 + 7) / 8, (BDIM * SDIM) / 16), 256, 0, stream>>>(
        keys_b, HDIM, uaw_b, HDIM, UaB, ua_keys, (long long)HDIM, HDIM, HDIM);

    const int vtiles = (VDIM + 15) / 16;   // 3142
    for (int t = 0; t < TDIM; ++t) {
        // q = h @ Wa^T + Wa_b : 128 x 512 x 512
        k_wmma_gemm<<<dim3((HDIM / 16 + 7) / 8, BDIM / 16), 256, 0, stream>>>(
            hb, HDIM, waw_b, HDIM, WaB, qbuf, (long long)HDIM, HDIM, HDIM);

        // attention + embed -> x (bf16), attn weights -> d_out
        k_attention<<<BDIM, 256, 0, stream>>>(qbuf, ua_keys, enc_out, VaW, VaB,
                                              emb, target, t, xbuf, out_attn);

        // gi = x @ Wih^T + Bih : 128 x 1536 x 1024
        k_wmma_gemm<<<dim3((3 * HDIM / 16 + 7) / 8, BDIM / 16), 256, 0, stream>>>(
            xbuf, 2 * HDIM, wih_b, 2 * HDIM, Bih, gibuf, (long long)(3 * HDIM), 3 * HDIM, 2 * HDIM);

        // gh = h @ Whh^T + Bhh : 128 x 1536 x 512
        k_wmma_gemm<<<dim3((3 * HDIM / 16 + 7) / 8, BDIM / 16), 256, 0, stream>>>(
            hb, HDIM, whh_b, HDIM, Bhh, ghbuf, (long long)(3 * HDIM), 3 * HDIM, HDIM);

        // GRU gate fusion -> new h (fp32 + bf16)
        k_gru<<<(BDIM * HDIM) / 256, 256, 0, stream>>>(gibuf, ghbuf, hbuf, hb);

        // logits = h_new @ out_w^T + out_b -> out_logp[:, t, :]
        // weight-streaming kernel: out_w read once per step from L2
        k_wmma_gemm_mblock<<<dim3((vtiles + 7) / 8, BDIM / 128), 256, 0, stream>>>(
            hb, HDIM, outw_b, HDIM, OutB,
            out_logp + (size_t)t * VDIM, (long long)TDIM * VDIM, VDIM, HDIM);

        // in-place log-softmax over V
        k_log_softmax<<<BDIM, 256, 0, stream>>>(out_logp, t);
    }

    // hT output
    k_copy_f32<<<(BDIM * HDIM) / 256, 256, 0, stream>>>(hbuf, out_hT, BDIM * HDIM);
}
